// TBiSegBlock_35278861369720
// MI455X (gfx1250) — matrix operand order repaired and checked by reading, not compiled, not run on hardware
//
#include <hip/hip_runtime.h>
#include <hip/hip_fp16.h>

typedef _Float16 h16;
typedef _Float16 v16h __attribute__((ext_vector_type(16)));
typedef _Float16 v8h  __attribute__((ext_vector_type(8)));
typedef float    v8f  __attribute__((ext_vector_type(8)));

union V16 { v16h v; v8h h[2]; };

__device__ __forceinline__ v8f wmma16(v16h a, v16h b, v8f c) {
  // v_wmma_f32_16x16x32_f16: D = A(16x32) * B(32x16) + C(16x16 f32)
  return __builtin_amdgcn_wmma_f32_16x16x32_f16(false, a, false, b, (short)0, c, false, false);
}

#define NTOK 25088   // 8*56*56
#define CDIM 256

// ---------------- weight fp32 -> fp16 ----------------
__global__ void cvt_f16_kernel(const float* __restrict__ src, h16* __restrict__ dst, int n) {
  int i = blockIdx.x * 256 + threadIdx.x;
  if (i < n) dst[i] = (h16)src[i];
}

// ---------------- LayerNorm over channels ----------------
template<bool CHANMAJOR>
__global__ __launch_bounds__(256)
void ln_kernel(const float* __restrict__ in, const float* __restrict__ g,
               const float* __restrict__ bt, h16* __restrict__ outv) {
  int t = blockIdx.x;
  int c = threadIdx.x;
  float v;
  if (CHANMAJOR) {
    int b = t / 3136, sp = t % 3136;
    v = in[((size_t)b * CDIM + c) * 3136 + sp];
  } else {
    v = in[(size_t)t * CDIM + c];
  }
  __shared__ float red[256];
  red[c] = v; __syncthreads();
  for (int s = 128; s > 0; s >>= 1) { if (c < s) red[c] += red[c + s]; __syncthreads(); }
  float mu = red[0] * (1.f / 256.f); __syncthreads();
  float d = v - mu;
  red[c] = d * d; __syncthreads();
  for (int s = 128; s > 0; s >>= 1) { if (c < s) red[c] += red[c + s]; __syncthreads(); }
  float var = red[0] * (1.f / 256.f);
  float y = d * rsqrtf(var + 1e-5f) * g[c] + bt[c];
  outv[(size_t)t * CDIM + c] = (h16)y;
}

// ---------------- generic WMMA GEMM: C[M][N] = A[M][K] * W[N][K]^T + bias ----------------
// MODE 0: -> f16 token-major            MODE 1: GELU -> f16 token-major
// MODE 2: + x(channel-major) -> f32 token-major (x1)
// MODE 3: + res(f32 token-major) -> f32 channel-major (d_out)
template<int MODE>
__global__ __launch_bounds__(256)
void gemm_wmma_kernel(const h16* __restrict__ A, int K,
                      const h16* __restrict__ W,
                      const float* __restrict__ bias,
                      void* __restrict__ out, int ldo,
                      const float* __restrict__ res) {
  int wave = threadIdx.x >> 5;
  int lane = threadIdx.x & 31;
  int half = lane >> 4;
  int l15  = lane & 15;
  int m0 = blockIdx.x * 128 + wave * 16;
  int n0 = blockIdx.y * 64;

  v8f acc[4] = {v8f{}, v8f{}, v8f{}, v8f{}};
  const h16* aptr = A + (size_t)(m0 + l15) * K + half * 16;
  for (int k0 = 0; k0 < K; k0 += 32) {
    V16 af;
    af.h[0] = *(const v8h*)(aptr + k0);
    af.h[1] = *(const v8h*)(aptr + k0 + 8);
#pragma unroll
    for (int f = 0; f < 4; ++f) {
      const h16* bp = W + (size_t)(n0 + f * 16 + l15) * K + k0 + half * 16;
      V16 bf;
      bf.h[0] = *(const v8h*)bp;
      bf.h[1] = *(const v8h*)(bp + 8);
      acc[f] = wmma16(af.v, bf.v, acc[f]);
    }
  }
#pragma unroll
  for (int f = 0; f < 4; ++f) {
    int n = n0 + f * 16 + l15;
    float bb = bias[n];
#pragma unroll
    for (int j = 0; j < 8; ++j) {
      int m = m0 + j + half * 8;
      float v = acc[f][j] + bb;
      if (MODE == 0) {
        ((h16*)out)[(size_t)m * ldo + n] = (h16)v;
      } else if (MODE == 1) {
        v = 0.5f * v * (1.f + erff(v * 0.70710678118654752f));
        ((h16*)out)[(size_t)m * ldo + n] = (h16)v;
      } else if (MODE == 2) {
        int b = m / 3136, sp = m % 3136;
        ((float*)out)[(size_t)m * ldo + n] = v + res[((size_t)b * CDIM + n) * 3136 + sp];
      } else {
        int b = m / 3136, sp = m % 3136;
        ((float*)out)[((size_t)b * CDIM + n) * 3136 + sp] = v + res[(size_t)m * CDIM + n];
      }
    }
  }
}

// ---------------- window mean-pool of q,k ----------------
__global__ __launch_bounds__(256)
void pool_kernel(const h16* __restrict__ qkv, float* __restrict__ pq, float* __restrict__ pk) {
  int br = blockIdx.x;              // b*49 + r
  int b = br / 49, r = br % 49;
  int wi = r / 7, wj = r % 7;
  int c = threadIdx.x;
  float sq = 0.f, sk = 0.f;
  for (int l = 0; l < 64; ++l) {
    int hh = wi * 8 + (l >> 3), ww = wj * 8 + (l & 7);
    size_t t = (size_t)b * 3136 + hh * 56 + ww;
    sq += (float)qkv[t * 768 + c];
    sk += (float)qkv[t * 768 + 256 + c];
  }
  pq[(size_t)br * 256 + c] = sq * (1.f / 64.f);
  pk[(size_t)br * 256 + c] = sk * (1.f / 64.f);
}

// ---------------- router scores + top-4 ----------------
__global__ __launch_bounds__(64)
void router_kernel(const float* __restrict__ pq, const float* __restrict__ pk, int* __restrict__ idx) {
  int b = blockIdx.x / 49, r = blockIdx.x % 49;
  __shared__ float sc[49];
  int kw = threadIdx.x;
  if (kw < 49) {
    const float* q = pq + ((size_t)b * 49 + r) * 256;
    const float* k = pk + ((size_t)b * 49 + kw) * 256;
    float s = 0.f;
    for (int c = 0; c < 256; ++c) s += q[c] * k[c];
    sc[kw] = s;
  }
  __syncthreads();
  if (threadIdx.x == 0) {
    for (int t = 0; t < 4; ++t) {
      int best = 0; float bv = -1e30f;
      for (int k2 = 0; k2 < 49; ++k2) if (sc[k2] > bv) { bv = sc[k2]; best = k2; }
      idx[((size_t)b * 49 + r) * 4 + t] = best;
      sc[best] = -1e30f;
    }
  }
}

// ---------------- v -> per-window d-major sequence layout ----------------
__global__ __launch_bounds__(256)
void vseq_kernel(const h16* __restrict__ qkv, h16* __restrict__ vs) {
  size_t t = blockIdx.x;
  int c = threadIdx.x;
  int b = (int)(t / 3136), sp = (int)(t % 3136);
  int hh = sp / 56, ww = sp % 56;
  int wi = hh >> 3, i = hh & 7, wj = ww >> 3, j = ww & 7;
  int r = wi * 7 + wj, l = i * 8 + j;
  int hd = c >> 5, d = c & 31;
  vs[((((size_t)b * 8 + hd) * 49 + r) * 32 + d) * 64 + l] = qkv[t * 768 + 512 + c];
}

// ---------------- gathered-window attention (WMMA QK^T + softmax + WMMA PV) ----------------
__global__ __launch_bounds__(128)
void attn_kernel(const h16* __restrict__ qkv, const h16* __restrict__ vs,
                 const int* __restrict__ idx, h16* __restrict__ o) {
  int bhr = blockIdx.x;                 // ((b*8)+h)*49 + r
  int r = bhr % 49; int bh = bhr / 49; int head = bh % 8; int b = bh / 8;
  int wi = r / 7, wj = r % 7;
  int wave = threadIdx.x >> 5, lane = threadIdx.x & 31;
  int half = lane >> 4, l15 = lane & 15;
  int l0 = wave * 16;

  __shared__ h16 probs[64 * 256];

  int idx4[4];
#pragma unroll
  for (int t = 0; t < 4; ++t) idx4[t] = idx[((size_t)b * 49 + r) * 4 + t];

  // ---- A fragment: q rows l0..l0+15, K = hd(32)
  int lrow = l0 + l15;
  int hh = wi * 8 + (lrow >> 3), ww = wj * 8 + (lrow & 7);
  size_t tq = (size_t)b * 3136 + hh * 56 + ww;
  const h16* qp = qkv + tq * 768 + head * 32 + half * 16;
  V16 aq;
  aq.h[0] = *(const v8h*)qp;
  aq.h[1] = *(const v8h*)(qp + 8);

  // ---- scores = q @ k^T over 4 gathered windows (M = 256 cols, 16 n-tiles)
  v8f sc[16];
#pragma unroll
  for (int nt = 0; nt < 16; ++nt) {
    int m = nt * 16 + l15;
    int tk = m >> 6, l2 = m & 63;
    int rw = idx4[tk];
    int wi2 = rw / 7, wj2 = rw % 7;
    int hh2 = wi2 * 8 + (l2 >> 3), ww2 = wj2 * 8 + (l2 & 7);
    size_t tkk = (size_t)b * 3136 + hh2 * 56 + ww2;
    const h16* kp = qkv + tkk * 768 + 256 + head * 32 + half * 16;
    V16 bk;
    bk.h[0] = *(const v8h*)kp;
    bk.h[1] = *(const v8h*)(kp + 8);
    v8f z = {};
    sc[nt] = wmma16(aq.v, bk.v, z);
  }

  // ---- softmax over 256 cols, per row (C layout: vgpr j = row, lane = col)
  const float scale = 0.0625f;   // 256^-0.5
  float rinv[8];
#pragma unroll
  for (int j = 0; j < 8; ++j) {
    float mx = -1e30f;
#pragma unroll
    for (int nt = 0; nt < 16; ++nt) mx = fmaxf(mx, sc[nt][j]);
    for (int s = 1; s < 16; s <<= 1) mx = fmaxf(mx, __shfl_xor(mx, s, 32));
    mx *= scale;
    float sm = 0.f;
#pragma unroll
    for (int nt = 0; nt < 16; ++nt) {
      float p = __expf(sc[nt][j] * scale - mx);
      sc[nt][j] = p;
      sm += p;
    }
    for (int s = 1; s < 16; s <<= 1) sm += __shfl_xor(sm, s, 32);
    rinv[j] = 1.f / sm;
  }

  // ---- stage probabilities to LDS (C layout -> row-major [l][m])
#pragma unroll
  for (int nt = 0; nt < 16; ++nt)
#pragma unroll
    for (int j = 0; j < 8; ++j) {
      int l = l0 + j + half * 8;
      int m = nt * 16 + l15;
      probs[l * 256 + m] = (h16)(sc[nt][j] * rinv[j]);
    }

  // ---- o = P(16x256) @ V(256x32), K-loop of 8x32
  v8f oacc[2] = {v8f{}, v8f{}};
  for (int kt = 0; kt < 8; ++kt) {
    const h16* pp = &probs[(l0 + l15) * 256 + kt * 32 + half * 16];
    V16 ap;
    ap.h[0] = *(const v8h*)pp;
    ap.h[1] = *(const v8h*)(pp + 8);
    int rw = idx4[kt >> 1];
    size_t vbase = (((size_t)bh * 49 + rw) * 32) * 64;
#pragma unroll
    for (int f = 0; f < 2; ++f) {
      int d = f * 16 + l15;
      const h16* vp = vs + vbase + (size_t)d * 64 + (kt & 1) * 32 + half * 16;
      V16 bv;
      bv.h[0] = *(const v8h*)vp;
      bv.h[1] = *(const v8h*)(vp + 8);
      oacc[f] = wmma16(ap.v, bv.v, oacc[f]);
    }
  }

  // ---- scatter o back to token-major [t][c]
#pragma unroll
  for (int f = 0; f < 2; ++f)
#pragma unroll
    for (int j = 0; j < 8; ++j) {
      int l = l0 + j + half * 8;
      int h2 = wi * 8 + (l >> 3), w2 = wj * 8 + (l & 7);
      size_t t = (size_t)b * 3136 + h2 * 56 + w2;
      o[t * 256 + head * 32 + f * 16 + l15] = (h16)oacc[f][j];
    }
}

// ---------------- depthwise 5x5 LePE + add attention output ----------------
__global__ __launch_bounds__(256)
void lepe_kernel(const h16* __restrict__ qkv, const float* __restrict__ lw,
                 const float* __restrict__ lb, const h16* __restrict__ o,
                 h16* __restrict__ ol) {
  size_t t = blockIdx.x;
  int c = threadIdx.x;
  int b = (int)(t / 3136), sp = (int)(t % 3136);
  int hh = sp / 56, ww = sp % 56;
  float acc = lb[c];
  for (int ky = 0; ky < 5; ++ky) {
    int y = hh + ky - 2;
    if (y < 0 || y >= 56) continue;
    for (int kx = 0; kx < 5; ++kx) {
      int x = ww + kx - 2;
      if (x < 0 || x >= 56) continue;
      acc += (float)qkv[((size_t)b * 3136 + y * 56 + x) * 768 + 512 + c] * lw[c * 25 + ky * 5 + kx];
    }
  }
  ol[t * 256 + c] = (h16)(acc + (float)o[t * 256 + c]);
}

// ============================================================================
extern "C" void kernel_launch(void* const* d_in, const int* in_sizes, int n_in,
                              void* d_out, int out_size, void* d_ws, size_t ws_size,
                              hipStream_t stream) {
  const float* x       = (const float*)d_in[0];
  const float* n1g     = (const float*)d_in[1];
  const float* n1b     = (const float*)d_in[2];
  const float* qkv_w   = (const float*)d_in[3];
  const float* qkv_b   = (const float*)d_in[4];
  const float* lepe_w  = (const float*)d_in[5];
  const float* lepe_b  = (const float*)d_in[6];
  const float* out_w   = (const float*)d_in[7];
  const float* out_b   = (const float*)d_in[8];
  const float* n2g     = (const float*)d_in[9];
  const float* n2b     = (const float*)d_in[10];
  const float* mlp_w1  = (const float*)d_in[11];
  const float* mlp_b1  = (const float*)d_in[12];
  const float* mlp_w2  = (const float*)d_in[13];
  const float* mlp_b2  = (const float*)d_in[14];
  float* out = (float*)d_out;

  // workspace arena
  char* base = (char*)d_ws;
  size_t off = 0;
  auto alloc = [&](size_t bytes) -> char* {
    off = (off + 255) & ~(size_t)255;
    char* p = base + off;
    off += bytes;
    return p;
  };
  h16*  wq  = (h16*)alloc(768 * 256 * 2);
  h16*  wo  = (h16*)alloc(256 * 256 * 2);
  h16*  w1  = (h16*)alloc(1024 * 256 * 2);
  h16*  w2  = (h16*)alloc(256 * 1024 * 2);
  float* pq = (float*)alloc(8 * 49 * 256 * 4);
  float* pk = (float*)alloc(8 * 49 * 256 * 4);
  int*  idb = (int*)alloc(8 * 49 * 4 * 4);
  float* x1 = (float*)alloc((size_t)NTOK * 256 * 4);
  h16*  xn  = (h16*)alloc((size_t)NTOK * 256 * 2);   // xn1 -> ol -> xn2 (reused)
  h16*  ob  = (h16*)alloc((size_t)NTOK * 256 * 2);
  h16*  vsb = (h16*)alloc((size_t)3136 * 2048 * 2);
  h16*  big = (h16*)alloc((size_t)NTOK * 1024 * 2);  // qkv (768 cols) then mlp hidden (1024)

  // 1) weights -> fp16
  cvt_f16_kernel<<<(768 * 256 + 255) / 256, 256, 0, stream>>>(qkv_w, wq, 768 * 256);
  cvt_f16_kernel<<<(256 * 256 + 255) / 256, 256, 0, stream>>>(out_w, wo, 256 * 256);
  cvt_f16_kernel<<<(1024 * 256 + 255) / 256, 256, 0, stream>>>(mlp_w1, w1, 1024 * 256);
  cvt_f16_kernel<<<(256 * 1024 + 255) / 256, 256, 0, stream>>>(mlp_w2, w2, 256 * 1024);

  // 2) LN1 (channel-major in -> token-major f16)
  ln_kernel<true><<<NTOK, 256, 0, stream>>>(x, n1g, n1b, xn);

  // 3) QKV GEMM: [25088,256] x [768,256]^T -> f16 [25088,768]
  gemm_wmma_kernel<0><<<dim3(NTOK / 128, 12), 256, 0, stream>>>(xn, 256, wq, qkv_b, big, 768, nullptr);

  // 4) window pooling of q,k
  pool_kernel<<<8 * 49, 256, 0, stream>>>(big, pq, pk);

  // 5) router + top-4
  router_kernel<<<8 * 49, 64, 0, stream>>>(pq, pk, idb);

  // 6) v -> d-major window sequence
  vseq_kernel<<<NTOK, 256, 0, stream>>>(big, vsb);

  // 7) gathered-window attention
  attn_kernel<<<8 * 8 * 49, 128, 0, stream>>>(big, vsb, idb, ob);

  // 8) LePE + add attention output (-> ol in xn region)
  lepe_kernel<<<NTOK, 256, 0, stream>>>(big, lepe_w, lepe_b, ob, xn);

  // 9) output projection + x residual -> x1 (f32 token-major)
  gemm_wmma_kernel<2><<<dim3(NTOK / 128, 4), 256, 0, stream>>>(xn, 256, wo, out_b, x1, 256, x);

  // 10) LN2 (token-major in)
  ln_kernel<false><<<NTOK, 256, 0, stream>>>(x1, n2g, n2b, xn);

  // 11) MLP1 + exact GELU -> f16 [25088,1024]
  gemm_wmma_kernel<1><<<dim3(NTOK / 128, 16), 256, 0, stream>>>(xn, 256, w1, mlp_b1, big, 1024, nullptr);

  // 12) MLP2 + x1 residual -> d_out (channel-major f32)
  gemm_wmma_kernel<3><<<dim3(NTOK / 128, 4), 256, 0, stream>>>(big, 1024, w2, mlp_b2, out, 256, x1);
}
